// QuantumNATHybrid_65481071398906
// MI455X (gfx1250) — compile-verified, matrix operand order checked by reading, not run on hardware
//
#include <hip/hip_runtime.h>
#include <hip/hip_bf16.h>

// ---------------------------------------------------------------------------
// Types for CDNA5 WMMA (wave32): V_WMMA_F32_16X16X32_F16
// ---------------------------------------------------------------------------
typedef __attribute__((ext_vector_type(16))) _Float16 v16h;
typedef __attribute__((ext_vector_type(8)))  _Float16 v8h;
typedef __attribute__((ext_vector_type(8)))  float    v8f;

union Frag16 {
    v16h     v;
    v8h      h[2];
    _Float16 e[16];
};

#define NB 4096               // batch

// workspace offsets (all 256B aligned)
#define OFF_H1    0ul                      // fp16 NHWC [B][28][28][8]  51,380,224 B
#define OFF_H2P   51380224ul               // fp16 NHWC [B][14][14][16] 25,690,112 B
#define OFF_H3P   77070336ul               // fp16 flat [B][1568]       12,845,056 B
#define OFF_M1    89915392ul               // fp32 [B][64]               1,048,576 B
#define OFF_M2    90963968ul               // fp32 [B][4]                   65,536 B
#define OFF_W2H   91029504ul               // fp16 [16][96]  reordered       3,072 B
#define OFF_W3H   91032576ul               // fp16 [32][160] reordered      10,240 B
#define OFF_WFH   91042816ul               // fp16 [64][1568]              200,704 B
#define OFF_STATS 91243520ul               // fp32 mu[4], var[4]

// branch-free 16B load with address clamp + data zeroing (v_cndmask/v_and)
__device__ __forceinline__ v8h masked_load16(const _Float16* __restrict__ p,
                                             unsigned msk) {
    union { v8h v; unsigned u[4]; } t;
    t.v = *(const v8h*)p;
    t.u[0] &= msk; t.u[1] &= msk; t.u[2] &= msk; t.u[3] &= msk;
    return t.v;
}

// ---------------------------------------------------------------------------
// K0: reorder + convert GEMM weights fp32 -> fp16 once per launch.
// Conv weights go to B[k][co]-friendly rows with k = tap*C + ci, zero padded.
// ---------------------------------------------------------------------------
__global__ void prep_weights(const float* __restrict__ c2w,
                             const float* __restrict__ c3w,
                             const float* __restrict__ f1w,
                             _Float16* __restrict__ w2h,
                             _Float16* __restrict__ w3h,
                             _Float16* __restrict__ wfh) {
    const int i = blockIdx.x * 256 + threadIdx.x;
    if (i < 16 * 96) {                       // conv2: [co][k], k = tap*8 + ci
        const int co = i / 96, k = i % 96;
        const int tap = k >> 3, ci = k & 7;
        w2h[i] = (tap < 9) ? (_Float16)c2w[co * 72 + ci * 9 + tap] : (_Float16)0.f;
    }
    if (i < 32 * 160) {                      // conv3: [co][k], k = tap*16 + ci
        const int co = i / 160, k = i % 160;
        const int tap = k >> 4, ci = k & 15;
        w3h[i] = (tap < 9) ? (_Float16)c3w[co * 144 + ci * 9 + tap] : (_Float16)0.f;
    }
    if (i < 100352) wfh[i] = (_Float16)f1w[i];
}

// ---------------------------------------------------------------------------
// K1: conv1 (Cin=1) + ReLU -> NHWC fp16, one thread per pixel (8 ch, 16B store)
// ---------------------------------------------------------------------------
__global__ void conv1_relu(const float* __restrict__ x,
                           const float* __restrict__ w,
                           const float* __restrict__ bias,
                           _Float16* __restrict__ h1) {
    const int i = blockIdx.x * 256 + threadIdx.x;       // over NB*784 pixels
    if (i >= NB * 784) return;
    const int px = i % 28;
    int t = i / 28;
    const int py = t % 28;
    const int b  = t / 28;

    float s[8];
#pragma unroll
    for (int co = 0; co < 8; ++co) s[co] = bias[co];

    const float* xim = x + b * 784;
#pragma unroll
    for (int ky = 0; ky < 3; ++ky) {
        const int iy = py + ky - 1;
        if (iy < 0 || iy >= 28) continue;
#pragma unroll
        for (int kx = 0; kx < 3; ++kx) {
            const int ix = px + kx - 1;
            if (ix < 0 || ix >= 28) continue;
            const float xv = xim[iy * 28 + ix];
#pragma unroll
            for (int co = 0; co < 8; ++co)
                s[co] += xv * w[co * 9 + ky * 3 + kx];
        }
    }
    union { v8h v; _Float16 e[8]; } o;
#pragma unroll
    for (int co = 0; co < 8; ++co) o.e[co] = (_Float16)fmaxf(s[co], 0.f);
    *(v8h*)(h1 + (size_t)i * 8) = o.v;
}

// ---------------------------------------------------------------------------
// K2: conv2 (8->16) + ReLU + 2x2 maxpool, implicit-GEMM on WMMA.
// NHWC input: each K=32 step = 4 taps x 8 ch -> 2 aligned b128 loads per lane.
// ---------------------------------------------------------------------------
__global__ void conv2_wmma(const _Float16* __restrict__ h1,
                           const _Float16* __restrict__ w2,
                           const float* __restrict__ b2,
                           _Float16* __restrict__ h2p) {
    const int wave = (blockIdx.x * blockDim.x + threadIdx.x) >> 5;
    const int lane = threadIdx.x & 31;
    const int m    = lane & 15;
    const int hi   = lane >> 4;

    const int b   = wave / 49;
    const int blk = wave % 49;              // 7x7 blocks of 4x4 pixels
    const int by  = (blk / 7) * 4;
    const int bx  = (blk % 7) * 4;
    const int py  = by + (m >> 2);
    const int px  = bx + (m & 3);

    const _Float16* h1b  = h1 + (size_t)b * 784 * 8;
    const _Float16* brow = w2 + m * 96;

    auto tapA = [&](int t) -> v8h {         // 8 channels of one shifted pixel
        const int ky = t / 3, kx = t % 3;
        const int iy = py + ky - 1, ix = px + kx - 1;
        const bool inb = (t < 9) & ((unsigned)iy < 28u) & ((unsigned)ix < 28u);
        const int cy = inb ? iy : 0, cx = inb ? ix : 0;
        return masked_load16(h1b + (size_t)(cy * 28 + cx) * 8,
                             inb ? 0xFFFFFFFFu : 0u);
    };

    v8f acc = {};
#pragma unroll
    for (int s = 0; s < 3; ++s) {
        Frag16 a, w;
        a.h[0] = tapA(4 * s + hi);          // k = 32s + {0..7}  (+8 for hi)
        a.h[1] = tapA(4 * s + 2 + hi);      // k = 32s + {16..23}(+8 for hi)
        const int kb = 32 * s + (hi ? 8 : 0);
        w.h[0] = *(const v8h*)(brow + kb);
        w.h[1] = *(const v8h*)(brow + kb + 16);
        acc = __builtin_amdgcn_wmma_f32_16x16x32_f16(false, a.v, false, w.v,
                                                     (short)0, acc, false, false);
    }
    // epilogue: +bias, ReLU, 2x2 maxpool (windows live inside acc regs)
    const float bias = b2[m];
    const float v0 = fmaxf(0.f, fmaxf(fmaxf(acc[0], acc[1]), fmaxf(acc[4], acc[5])) + bias);
    const float v1 = fmaxf(0.f, fmaxf(fmaxf(acc[2], acc[3]), fmaxf(acc[6], acc[7])) + bias);
    const int gy  = (by >> 1) + hi;
    const int gx  = (bx >> 1);
    const int pix = (b * 14 + gy) * 14 + gx;
    h2p[(size_t)pix * 16 + m]        = (_Float16)v0;   // NHWC out
    h2p[(size_t)(pix + 1) * 16 + m]  = (_Float16)v1;
}

// ---------------------------------------------------------------------------
// K3: conv3 (16->32) + ReLU + 2x2 maxpool on WMMA.
// NHWC input: each K=32 step = 2 taps x 16 ch; lane takes channel-half hi*8.
// A fragments hoisted and reused across the two N-tiles.
// ---------------------------------------------------------------------------
__global__ void conv3_wmma(const _Float16* __restrict__ h2p,
                           const _Float16* __restrict__ w3,
                           const float* __restrict__ b3,
                           _Float16* __restrict__ h3p) {
    const int wave = (blockIdx.x * blockDim.x + threadIdx.x) >> 5;
    const int lane = threadIdx.x & 31;
    const int m = lane & 15, hi = lane >> 4;

    const int b   = wave >> 4;
    const int blk = wave & 15;              // 4x4 blocks of 4x4 pixels over 14x14
    const int by  = (blk >> 2) * 4;
    const int bx  = (blk & 3) * 4;
    const int py  = by + (m >> 2);
    const int px  = bx + (m & 3);

    const _Float16* h2b = h2p + (size_t)b * 196 * 16;
    const int choff = hi * 8;

    auto tapA3 = [&](int t) -> v8h {        // channel-half of one shifted pixel
        const int ky = t / 3, kx = t % 3;   // t is compile-time const after unroll
        const int iy = py + ky - 1, ix = px + kx - 1;
        const bool inb = (t < 9) & ((unsigned)iy < 14u) & ((unsigned)ix < 14u);
        const int cy = inb ? iy : 0, cx = inb ? ix : 0;
        return masked_load16(h2b + (size_t)(cy * 14 + cx) * 16 + choff,
                             inb ? 0xFFFFFFFFu : 0u);
    };

    Frag16 a[5];
#pragma unroll
    for (int s = 0; s < 5; ++s) {
        a[s].h[0] = tapA3(2 * s);           // k = 32s + (hi?8:0) + {0..7}
        a[s].h[1] = tapA3(2 * s + 1);       // k = 32s + (hi?8:0) + {16..23}
    }

#pragma unroll
    for (int cot = 0; cot < 2; ++cot) {
        const _Float16* brow = w3 + (size_t)(16 * cot + m) * 160;
        v8f acc = {};
#pragma unroll
        for (int s = 0; s < 5; ++s) {
            Frag16 w;
            const int kb = 32 * s + (hi ? 8 : 0);
            w.h[0] = *(const v8h*)(brow + kb);
            w.h[1] = *(const v8h*)(brow + kb + 16);
            acc = __builtin_amdgcn_wmma_f32_16x16x32_f16(false, a[s].v, false, w.v,
                                                         (short)0, acc, false, false);
        }
        const int n = 16 * cot + m;
        const float bias = b3[n];
        const float v0 = fmaxf(0.f, fmaxf(fmaxf(acc[0], acc[1]), fmaxf(acc[4], acc[5])) + bias);
        const float v1 = fmaxf(0.f, fmaxf(fmaxf(acc[2], acc[3]), fmaxf(acc[6], acc[7])) + bias);
        const int gy  = (by >> 1) + hi;
        const int gx0 = (bx >> 1), gx1 = gx0 + 1;
        if (gy < 7) {                       // output flat NCHW [b][c*49+y*7+x]
            _Float16* dst = h3p + (size_t)b * 1568 + n * 49 + gy * 7;
            if (gx0 < 7) dst[gx0] = (_Float16)v0;
            if (gx1 < 7) dst[gx1] = (_Float16)v1;
        }
    }
}

// ---------------------------------------------------------------------------
// K4: fc1 GEMM [4096,1568]x[1568,64] + bias + ReLU on WMMA. K = 49*32 exactly.
// ---------------------------------------------------------------------------
__global__ void fc1_wmma(const _Float16* __restrict__ h3p,
                         const _Float16* __restrict__ wf,
                         const float* __restrict__ fb,
                         float* __restrict__ m1) {
    const int wave = (blockIdx.x * blockDim.x + threadIdx.x) >> 5;
    const int lane = threadIdx.x & 31;
    const int m = lane & 15, hi = lane >> 4;
    const int mt = wave >> 2, nt = wave & 3;

    const _Float16* arow = h3p + (size_t)(16 * mt + m) * 1568;
    const _Float16* brow = wf  + (size_t)(16 * nt + m) * 1568;
    const int off = hi ? 8 : 0;

    v8f acc = {};
    for (int s = 0; s < 49; ++s) {
        const int kb = 32 * s + off;
        __builtin_prefetch(arow + kb + 128, 0, 1);   // global_prefetch_b8
        Frag16 a, w;
        a.h[0] = *(const v8h*)(arow + kb);
        a.h[1] = *(const v8h*)(arow + kb + 16);
        w.h[0] = *(const v8h*)(brow + kb);
        w.h[1] = *(const v8h*)(brow + kb + 16);
        acc = __builtin_amdgcn_wmma_f32_16x16x32_f16(false, a.v, false, w.v,
                                                     (short)0, acc, false, false);
    }
    const int nn = 16 * nt + m;
    const float bias = fb[nn];
#pragma unroll
    for (int r = 0; r < 8; ++r) {
        const int mm = 16 * mt + r + 8 * hi;
        m1[mm * 64 + nn] = fmaxf(0.f, acc[r] + bias);
    }
}

// ---------------------------------------------------------------------------
// K5: fc2 [4096,64]x[64,4] scalar (tiny)
// ---------------------------------------------------------------------------
__global__ void fc2_kernel(const float* __restrict__ m1,
                           const float* __restrict__ w,
                           const float* __restrict__ bias,
                           float* __restrict__ m2) {
    const int i = blockIdx.x * 256 + threadIdx.x;
    if (i >= NB * 4) return;
    const int j = i & 3, b = i >> 2;
    float s = bias[j];
    const float* row = m1 + b * 64;
#pragma unroll
    for (int k = 0; k < 64; ++k) s += row[k] * w[j * 64 + k];
    m2[i] = s;
}

// ---------------------------------------------------------------------------
// K6: deterministic batch-norm stats (one block per column, tree reduction)
// ---------------------------------------------------------------------------
__global__ void bn_stats(const float* __restrict__ m2, float* __restrict__ stats) {
    __shared__ float ssum[256], ssq[256];
    const int j = blockIdx.x, t = threadIdx.x;
    float s = 0.f, q = 0.f;
    for (int b = t; b < NB; b += 256) {
        const float v = m2[b * 4 + j];
        s += v; q += v * v;
    }
    ssum[t] = s; ssq[t] = q;
    __syncthreads();
    for (int w = 128; w > 0; w >>= 1) {
        if (t < w) { ssum[t] += ssum[t + w]; ssq[t] += ssq[t + w]; }
        __syncthreads();
    }
    if (t == 0) {
        const float mu = ssum[0] / (float)NB;
        stats[j]     = mu;
        stats[4 + j] = ssq[0] / (float)NB - mu * mu;   // biased var, as jnp.var
    }
}

// ---------------------------------------------------------------------------
// K7: BN apply + 4-qubit statevector circuit + cls head, one sample per thread
// ---------------------------------------------------------------------------
__global__ void finalize(const float* __restrict__ m2,
                         const float* __restrict__ stats,
                         const float* __restrict__ bng, const float* __restrict__ bnb,
                         const float* __restrict__ qw,
                         const float* __restrict__ c1w, const float* __restrict__ c1b,
                         const float* __restrict__ c2w, const float* __restrict__ c2b,
                         float* __restrict__ out) {
    const int b = blockIdx.x * 256 + threadIdx.x;
    if (b >= NB) return;

    float theta[4];
#pragma unroll
    for (int j = 0; j < 4; ++j) {
        const float mu = stats[j], var = stats[4 + j];
        theta[j] = (m2[b * 4 + j] - mu) * rsqrtf(var + 1e-5f) * bng[j] + bnb[j];
    }

    // statevector: wire 0 is MSB of the 4-bit index (shape (2,2,2,2))
    float re[16], im[16];
#pragma unroll
    for (int i = 0; i < 16; ++i) { re[i] = 0.f; im[i] = 0.f; }
    re[0] = 1.f;

#pragma unroll
    for (int w = 0; w < 4; ++w) {           // RY(theta_w), real rotation
        const int mask = 1 << (3 - w);
        const float c = cosf(theta[w] * 0.5f), s = sinf(theta[w] * 0.5f);
#pragma unroll
        for (int idx = 0; idx < 16; ++idx) {
            if (idx & mask) continue;
            const int i1 = idx | mask;
            const float r0 = re[idx], q0 = im[idx], r1 = re[i1], q1 = im[i1];
            re[idx] = c * r0 - s * r1;  im[idx] = c * q0 - s * q1;
            re[i1]  = s * r0 + c * r1;  im[i1]  = s * q0 + c * q1;
        }
    }

#pragma unroll
    for (int l = 0; l < 2; ++l) {           // StronglyEntanglingLayers
#pragma unroll
        for (int w = 0; w < 4; ++w) {
            const float* p = qw + (l * 4 + w) * 3;
            const float phi = p[0], th = p[1], om = p[2];
            const float ct = cosf(th * 0.5f), st = sinf(th * 0.5f);
            const float aa = (phi + om) * 0.5f, bb = (phi - om) * 0.5f;
            const float ca = cosf(aa), sa = sinf(aa);
            const float cb = cosf(bb), sb = sinf(bb);
            const float u00r =  ca * ct, u00i = -sa * ct;
            const float u01r = -cb * st, u01i = -sb * st;
            const float u10r =  cb * st, u10i = -sb * st;
            const float u11r =  ca * ct, u11i =  sa * ct;
            const int mask = 1 << (3 - w);
#pragma unroll
            for (int idx = 0; idx < 16; ++idx) {
                if (idx & mask) continue;
                const int i1 = idx | mask;
                const float r0 = re[idx], q0 = im[idx], r1 = re[i1], q1 = im[i1];
                re[idx] = u00r * r0 - u00i * q0 + u01r * r1 - u01i * q1;
                im[idx] = u00r * q0 + u00i * r0 + u01r * q1 + u01i * r1;
                re[i1]  = u10r * r0 - u10i * q0 + u11r * r1 - u11i * q1;
                im[i1]  = u10r * q0 + u10i * r0 + u11r * q1 + u11i * r1;
            }
        }
        const int r = (l % 3) + 1;
#pragma unroll
        for (int i = 0; i < 4; ++i) {       // CNOT ring
            const int tq = (i + r) & 3;
            const int cm = 1 << (3 - i), tm = 1 << (3 - tq);
#pragma unroll
            for (int idx = 0; idx < 16; ++idx) {
                if ((idx & cm) && !(idx & tm)) {
                    const int j1 = idx | tm;
                    float tr = re[idx]; re[idx] = re[j1]; re[j1] = tr;
                    float ti = im[idx]; im[idx] = im[j1]; im[j1] = ti;
                }
            }
        }
    }

    float z[4];
#pragma unroll
    for (int w = 0; w < 4; ++w) {           // <Z_w>
        const int mask = 1 << (3 - w);
        float acc = 0.f;
#pragma unroll
        for (int idx = 0; idx < 16; ++idx) {
            const float p = re[idx] * re[idx] + im[idx] * im[idx];
            acc += (idx & mask) ? -p : p;
        }
        z[w] = acc;
    }

    float h16[16];
#pragma unroll
    for (int j = 0; j < 16; ++j) {
        float s = c1b[j];
#pragma unroll
        for (int k = 0; k < 4; ++k) s += z[k] * c1w[j * 4 + k];
        h16[j] = fmaxf(s, 0.f);
    }
#pragma unroll
    for (int o = 0; o < 4; ++o) {
        float s = c2b[o];
#pragma unroll
        for (int j = 0; j < 16; ++j) s += h16[j] * c2w[o * 16 + j];
        out[b * 4 + o] = s;
    }
}

// ---------------------------------------------------------------------------
// launch
// ---------------------------------------------------------------------------
extern "C" void kernel_launch(void* const* d_in, const int* in_sizes, int n_in,
                              void* d_out, int out_size, void* d_ws, size_t ws_size,
                              hipStream_t stream) {
    (void)in_sizes; (void)n_in; (void)out_size; (void)ws_size;

    const float* x    = (const float*)d_in[0];
    const float* c1w  = (const float*)d_in[1];
    const float* c1b  = (const float*)d_in[2];
    const float* c2w  = (const float*)d_in[3];
    const float* c2b  = (const float*)d_in[4];
    const float* c3w  = (const float*)d_in[5];
    const float* c3b  = (const float*)d_in[6];
    const float* f1w  = (const float*)d_in[7];
    const float* f1b  = (const float*)d_in[8];
    const float* f2w  = (const float*)d_in[9];
    const float* f2b  = (const float*)d_in[10];
    const float* bng  = (const float*)d_in[11];
    const float* bnb  = (const float*)d_in[12];
    const float* qw   = (const float*)d_in[13];
    const float* cl1w = (const float*)d_in[14];
    const float* cl1b = (const float*)d_in[15];
    const float* cl2w = (const float*)d_in[16];
    const float* cl2b = (const float*)d_in[17];

    char* ws = (char*)d_ws;
    _Float16* h1    = (_Float16*)(ws + OFF_H1);
    _Float16* h2p   = (_Float16*)(ws + OFF_H2P);
    _Float16* h3p   = (_Float16*)(ws + OFF_H3P);
    float*    m1    = (float*)   (ws + OFF_M1);
    float*    m2    = (float*)   (ws + OFF_M2);
    _Float16* w2h   = (_Float16*)(ws + OFF_W2H);
    _Float16* w3h   = (_Float16*)(ws + OFF_W3H);
    _Float16* wfh   = (_Float16*)(ws + OFF_WFH);
    float*    stats = (float*)   (ws + OFF_STATS);

    prep_weights<<<392, 256, 0, stream>>>(c2w, c3w, f1w, w2h, w3h, wfh);
    conv1_relu  <<<(NB * 784) / 256, 256, 0, stream>>>(x, c1w, c1b, h1);
    conv2_wmma  <<<(NB * 49) / 8, 256, 0, stream>>>(h1, w2h, c2b, h2p);   // 8 waves/blk
    conv3_wmma  <<<(NB * 16) / 8, 256, 0, stream>>>(h2p, w3h, c3b, h3p);
    fc1_wmma    <<<(256 * 4) / 8, 256, 0, stream>>>(h3p, wfh, f1b, m1);
    fc2_kernel  <<<(NB * 4) / 256, 256, 0, stream>>>(m1, f2w, f2b, m2);
    bn_stats    <<<4, 256, 0, stream>>>(m2, stats);
    finalize    <<<NB / 256, 256, 0, stream>>>(m2, stats, bng, bnb, qw,
                                               cl1w, cl1b, cl2w, cl2b, (float*)d_out);
}
